// QuantumTextClassifier_65481071400026
// MI455X (gfx1250) — matrix operand order repaired and checked by reading, not compiled, not run on hardware
//
#include <hip/hip_runtime.h>
#include <hip/hip_bf16.h>
#include <math.h>

#define NUM_BLOCKS 4
#define EDIM 128
#define TDIM 2048
#define BDIM 32
#define CDIM 4
#define GROUPS 16              // blocks per batch-row in the fused kernel
#define LN_EPS 1e-5f

typedef __attribute__((ext_vector_type(16))) _Float16 v16h;
typedef __attribute__((ext_vector_type(8)))  float    v8f;

// ---------------------------------------------------------------------------
// LayerNorm over the 128-wide feature dim held as 4 floats/lane (wave32).
// Butterfly reduction keeps everything in VGPRs; no LDS, no barriers.
// ---------------------------------------------------------------------------
__device__ __forceinline__ void ln128(float s[4], const float4 g, const float4 bb) {
    float sum = s[0] + s[1] + s[2] + s[3];
    float sq  = s[0]*s[0] + s[1]*s[1] + s[2]*s[2] + s[3]*s[3];
#pragma unroll
    for (int off = 16; off >= 1; off >>= 1) {
        sum += __shfl_xor(sum, off, 32);
        sq  += __shfl_xor(sq,  off, 32);
    }
    const float mean = sum * (1.0f / 128.0f);
    const float var  = sq  * (1.0f / 128.0f) - mean * mean;
    const float r    = rsqrtf(var + LN_EPS);
    s[0] = (s[0] - mean) * r * g.x + bb.x;
    s[1] = (s[1] - mean) * r * g.y + bb.y;
    s[2] = (s[2] - mean) * r * g.z + bb.z;
    s[3] = (s[3] - mean) * r * g.w + bb.w;
}

__global__ __launch_bounds__(256) void qtc_zero(float* __restrict__ p, int n) {
    int i = blockIdx.x * blockDim.x + threadIdx.x;
    if (i < n) p[i] = 0.0f;
}

// ---------------------------------------------------------------------------
// Fused: gather + PE + 4x(cumprod-cos scan, LN, LN) + partial T-mean.
// One wave owns one (b,t) row at a time: e = 4*lane + j.
// grid = (GROUPS, B), block = 256 (8 waves); each wave processes 16 rows.
// ---------------------------------------------------------------------------
__global__ __launch_bounds__(256) void qtc_fused_rows(
    const int*   __restrict__ tokens, const float* __restrict__ emb,
    const float* __restrict__ g1, const float* __restrict__ b1,
    const float* __restrict__ g2, const float* __restrict__ b2,
    float* __restrict__ pool)                 // [B, E] accumulator (pre-zeroed)
{
    const int b    = blockIdx.y;
    const int lane = threadIdx.x & 31;
    const int wave = threadIdx.x >> 5;
    constexpr int wavesPerWG = 8;             // 256 threads
    constexpr int rowsPerWG  = TDIM / GROUPS; // 128
    const int t0 = blockIdx.x * rowsPerWG;

    // LN params are uniform across rows: hoist into registers once per wave.
    float4 G1[NUM_BLOCKS], B1[NUM_BLOCKS], G2[NUM_BLOCKS], B2[NUM_BLOCKS];
#pragma unroll
    for (int i = 0; i < NUM_BLOCKS; ++i) {
        G1[i] = *(const float4*)(g1 + i * EDIM + lane * 4);
        B1[i] = *(const float4*)(b1 + i * EDIM + lane * 4);
        G2[i] = *(const float4*)(g2 + i * EDIM + lane * 4);
        B2[i] = *(const float4*)(b2 + i * EDIM + lane * 4);
    }

    // Sinusoidal PE frequencies: e = 4*lane+j, pair p = e>>1.
    const float kLn = -logf(10000.0f) / (float)EDIM;             // folded at compile time
    const float f0  = __expf((float)(2 * (2 * lane    )) * kLn); // pairs (j=0,1)
    const float f1  = __expf((float)(2 * (2 * lane + 1)) * kLn); // pairs (j=2,3)

    float acc[4] = {0.f, 0.f, 0.f, 0.f};

    for (int t = t0 + wave; t < t0 + rowsPerWG; t += wavesPerWG) {
        const int tok = tokens[b * TDIM + t];
        const float4 xr = *(const float4*)(emb + (size_t)tok * EDIM + lane * 4);
        float x[4] = {xr.x, xr.y, xr.z, xr.w};
        const float tp = (float)t;
        x[0] += __sinf(tp * f0);  x[1] += __cosf(tp * f0);
        x[2] += __sinf(tp * f1);  x[3] += __cosf(tp * f1);

#pragma unroll
        for (int i = 0; i < NUM_BLOCKS; ++i) {
            // q = cumprod(cos(x)) along the 128-wide axis: local prefix + wave scan
            const float c0 = __cosf(x[0]), c1 = __cosf(x[1]);
            const float c2 = __cosf(x[2]), c3 = __cosf(x[3]);
            const float p0 = c0, p1 = p0 * c1, p2 = p1 * c2, p3 = p2 * c3;
            float scan = p3;                         // inclusive scan of lane totals
#pragma unroll
            for (int off = 1; off < 32; off <<= 1) {
                const float o = __shfl_up(scan, off, 32);
                if (lane >= off) scan *= o;
            }
            float excl = __shfl_up(scan, 1, 32);
            if (lane == 0) excl = 1.0f;
            const float q[4] = {excl * p0, excl * p1, excl * p2, excl * p3};

            float s[4] = {x[0] + q[0], x[1] + q[1], x[2] + q[2], x[3] + q[3]};
            ln128(s, G1[i], B1[i]);
            float s2[4] = {s[0] + q[0], s[1] + q[1], s[2] + q[2], s[3] + q[3]};
            ln128(s2, G2[i], B2[i]);
            x[0] = s2[0]; x[1] = s2[1]; x[2] = s2[2]; x[3] = s2[3];
        }
        acc[0] += x[0]; acc[1] += x[1]; acc[2] += x[2]; acc[3] += x[3];
    }

    float* dst = pool + b * EDIM + lane * 4;
    atomicAdd(dst + 0, acc[0]);
    atomicAdd(dst + 1, acc[1]);
    atomicAdd(dst + 2, acc[2]);
    atomicAdd(dst + 3, acc[3]);
}

// ---------------------------------------------------------------------------
// Final projection (pool/T) @ W + b via V_WMMA_F32_16X16X32_F16.
// 2 waves: wave w owns output rows [16w, 16w+16). K-loop: 4 x (K=32) over E=128.
// N padded 4 -> 16 with a multiplicative zero-mask on B (no divergent loads).
// ISA 7.12.2 VGPR layouts.
// ---------------------------------------------------------------------------
__global__ __launch_bounds__(64) void qtc_proj_wmma(
    const float* __restrict__ pool, const float* __restrict__ W,
    const float* __restrict__ bias, float* __restrict__ out)
{
    const int lane   = threadIdx.x & 31;
    const int waveId = threadIdx.x >> 5;   // 0 or 1
    const int half   = lane >> 4;          // lane group (0: lanes 0-15, 1: 16-31)
    const int nl     = lane & 15;
    const float invT  = 1.0f / (float)TDIM;
    const float nmask = (nl < CDIM) ? 1.0f : 0.0f;  // N-padding mask (no branches)
    const int   ncol  = nl & (CDIM - 1);            // clamped, always in-bounds

    v8f c = {};
#pragma unroll
    for (int kb = 0; kb < EDIM; kb += 32) {
        // A (16x32 f16): lane holds row M = nl. ISA layout per lane-half:
        //   idx 0..7  -> K = half*8 + idx       (contiguous 8 floats)
        //   idx 8..15 -> K = 16 + half*8 + idx-8 (contiguous 8 floats)
        const int row = waveId * 16 + nl;
        const float* arow = pool + row * EDIM + kb + half * 8;
        const float4 a0 = *(const float4*)(arow + 0);
        const float4 a1 = *(const float4*)(arow + 4);
        const float4 a2 = *(const float4*)(arow + 16);
        const float4 a3 = *(const float4*)(arow + 20);
        v16h a;
        a[0]  = (_Float16)(a0.x * invT); a[1]  = (_Float16)(a0.y * invT);
        a[2]  = (_Float16)(a0.z * invT); a[3]  = (_Float16)(a0.w * invT);
        a[4]  = (_Float16)(a1.x * invT); a[5]  = (_Float16)(a1.y * invT);
        a[6]  = (_Float16)(a1.z * invT); a[7]  = (_Float16)(a1.w * invT);
        a[8]  = (_Float16)(a2.x * invT); a[9]  = (_Float16)(a2.y * invT);
        a[10] = (_Float16)(a2.z * invT); a[11] = (_Float16)(a2.w * invT);
        a[12] = (_Float16)(a3.x * invT); a[13] = (_Float16)(a3.y * invT);
        a[14] = (_Float16)(a3.z * invT); a[15] = (_Float16)(a3.w * invT);

        // B (32x16 f16): lane holds col N = nl; same K order as A:
        // idx 0..7 -> K = half*8 + idx, idx 8..15 -> K = 16 + half*8 + idx-8. Unconditional clamped loads + mask.
        v16h bm;
#pragma unroll
        for (int kk = 0; kk < 16; ++kk) {
            const int k = (kk < 8) ? (half * 8 + kk) : (16 + half * 8 + (kk - 8));
            bm[kk] = (_Float16)(W[(kb + k) * CDIM + ncol] * nmask);
        }
        c = __builtin_amdgcn_wmma_f32_16x16x32_f16(
                /*neg_a=*/false, a, /*neg_b=*/false, bm,
                /*c_mod=*/(short)0, c, /*reuse_a=*/false, /*reuse_b=*/false);
    }

    // C/D layout: VGPR v -> M = v + 8*half, N = nl.
    const float bv = bias[ncol];               // unconditional clamped load
#pragma unroll
    for (int v = 0; v < 8; ++v) {
        const int grow = waveId * 16 + half * 8 + v;
        if (nl < CDIM) out[grow * CDIM + nl] = c[v] + bv;
    }
}

// ---------------------------------------------------------------------------
extern "C" void kernel_launch(void* const* d_in, const int* in_sizes, int n_in,
                              void* d_out, int out_size, void* d_ws, size_t ws_size,
                              hipStream_t stream) {
    (void)in_sizes; (void)n_in; (void)out_size; (void)ws_size;
    const int*   tokens = (const int*)  d_in[0];
    const float* emb    = (const float*)d_in[1];
    const float* g1     = (const float*)d_in[2];
    const float* b1     = (const float*)d_in[3];
    const float* g2     = (const float*)d_in[4];
    const float* b2     = (const float*)d_in[5];
    // d_in[6] = q_weights: provably no effect on the output (see reference comment)
    const float* W      = (const float*)d_in[7];
    const float* bias   = (const float*)d_in[8];
    float* out  = (float*)d_out;
    float* pool = (float*)d_ws;                  // [BDIM * EDIM] f32 accumulator

    qtc_zero<<<dim3((BDIM * EDIM + 255) / 256), 256, 0, stream>>>(pool, BDIM * EDIM);
    qtc_fused_rows<<<dim3(GROUPS, BDIM), 256, 0, stream>>>(tokens, emb, g1, b1, g2, b2, pool);
    qtc_proj_wmma<<<dim3(1), 64, 0, stream>>>(pool, W, bias, out);
}
